// Half_Graph_41927470743596
// MI455X (gfx1250) — matrix operand order repaired and checked by reading, not compiled, not run on hardware
//
#include <hip/hip_runtime.h>
#include <hip/hip_bf16.h>
#include <math.h>

typedef __attribute__((ext_vector_type(16))) _Float16 v16h;
typedef __attribute__((ext_vector_type(8)))  _Float16 v8h;
typedef __attribute__((ext_vector_type(8)))  float    v8f;

#define NB 8
#define NC 10
#define NH 192
#define NW 192
#define HW   (NH*NW)          /* 36864   */
#define BCHW (NB*NC*HW)       /* 2949120 */
#define BHW  (NB*HW)          /* 294912  */

/* d_out offsets (floats): xh_new[3,B,10,H,W] | decomp_map | comp_map_u | comp_map_l */
#define OUT_DMAP (3*BCHW)
#define OUT_CMU  (OUT_DMAP + NB*3*HW)
#define OUT_CML  (OUT_CMU + BHW)

/* ws offsets (floats) */
#define WS_MSG_U 0
#define WS_MSG_L BCHW
#define WS_ATT1  (2*BCHW)
#define WS_ATT2  (2*BCHW + BHW)

/* ------------------------------------------------------------------ */
/* Kernel 1: attention / composition maps + softmax + slot-0 copy      */
/* ------------------------------------------------------------------ */
__global__ __launch_bounds__(256) void fuse_maps_kernel(
    const float* __restrict__ f1, const float* __restrict__ h0,
    const float* __restrict__ h1, const float* __restrict__ h2,
    const float* __restrict__ pn,
    const float* __restrict__ w_dmap, const float* __restrict__ b_dmap,
    const float* __restrict__ w_cau,  const float* __restrict__ b_cau,
    const float* __restrict__ w_cal,  const float* __restrict__ b_cal,
    float* __restrict__ outp, float* __restrict__ ws)
{
    __shared__ float s_wd[90], s_bd[3], s_wu[40], s_wl[20], s_bul[2];
    int tid = threadIdx.x;
    if (tid < 90) s_wd[tid] = w_dmap[tid];
    if (tid < 3)  s_bd[tid] = b_dmap[tid];
    if (tid >= 128 && tid < 168) s_wu[tid-128] = w_cau[tid-128];
    if (tid >= 168 && tid < 188) s_wl[tid-168] = w_cal[tid-168];
    if (tid == 200) s_bul[0] = b_cau[0];
    if (tid == 201) s_bul[1] = b_cal[0];
    __syncthreads();

    int pix = blockIdx.x * 256 + tid;        /* [0, BHW) */
    int b  = pix / HW;
    int hw = pix - b*HW;
    size_t base = (size_t)b*NC*HW + hw;

    float f1v[10], h1v[10], h2v[10];
#pragma unroll
    for (int c = 0; c < 10; ++c) {
        f1v[c] = f1[base + (size_t)c*HW];
        h1v[c] = h1[base + (size_t)c*HW];
        h2v[c] = h2[base + (size_t)c*HW];
    }

    float dm[3];
#pragma unroll
    for (int o = 0; o < 3; ++o) {
        float s = s_bd[o];
#pragma unroll
        for (int c = 0; c < 10; ++c) {
            s += s_wd[o*30 +      c] * f1v[c];
            s += s_wd[o*30 + 10 + c] * h1v[c];
            s += s_wd[o*30 + 20 + c] * h2v[c];
        }
        dm[o] = s;
        outp[OUT_DMAP + ((size_t)b*3 + o)*HW + hw] = s;
    }
    float mx = fmaxf(dm[0], fmaxf(dm[1], dm[2]));
    float e0 = __expf(dm[0]-mx), e1 = __expf(dm[1]-mx), e2 = __expf(dm[2]-mx);
    float inv = 1.0f / (e0 + e1 + e2);
    ws[WS_ATT1 + (size_t)b*HW + hw] = e1 * inv;
    ws[WS_ATT2 + (size_t)b*HW + hw] = e2 * inv;

    float su = s_bul[0];
#pragma unroll
    for (int i = 0; i < 4; ++i) {
        const float* p = pn + (size_t)(i+1)*BCHW;
#pragma unroll
        for (int c = 0; c < 10; ++c) su += s_wu[i*10 + c] * p[base + (size_t)c*HW];
    }
    outp[OUT_CMU + (size_t)b*HW + hw] = 1.0f / (1.0f + __expf(-su));

    float sl = s_bul[1];
#pragma unroll
    for (int i = 0; i < 2; ++i) {
        const float* p = pn + (size_t)(i+5)*BCHW;
#pragma unroll
        for (int c = 0; c < 10; ++c) sl += s_wl[i*10 + c] * p[base + (size_t)c*HW];
    }
    outp[OUT_CML + (size_t)b*HW + hw] = 1.0f / (1.0f + __expf(-sl));

    /* xh_new[0] = h_nodes[0] */
#pragma unroll
    for (int c = 0; c < 10; ++c)
        outp[base + (size_t)c*HW] = h0[base + (size_t)c*HW];
}

/* ------------------------------------------------------------------ */
/* Kernel 2: 3x3 conv 20->10 + BN + ReLU as 9 tap-GEMMs on WMMA        */
/*   input ch 0..9  = inA * (modA ? modA : 1)                          */
/*   input ch 10..19= inB * (modB ? modB : 1)                          */
/*   outp (+)= relu(gam*conv + bet)                                    */
/*   LDS is channel-contiguous so every fragment is a ds_load_b128.    */
/* ------------------------------------------------------------------ */
__global__ __launch_bounds__(128) void conv3x3_wmma_kernel(
    const float* __restrict__ inA, const float* __restrict__ modA,
    const float* __restrict__ inB, const float* __restrict__ modB,
    const float* __restrict__ wgt, const float* __restrict__ gam,
    const float* __restrict__ bet, float* __restrict__ outp, int accumulate)
{
    __shared__ __align__(16) _Float16 sIn[3][68][32];  /* [row][x(-1..64)][c pad 32] */
    __shared__ __align__(16) _Float16 sW[9][16][32];   /* [tap][outch][c pad 32]     */

    int tid  = threadIdx.x;
    int bx   = blockIdx.x;
    int xblk = bx % 3;
    int y    = (bx / 3) % NH;
    int b    = bx / (3*NH);
    int x0   = xblk * 64;

    /* prefetch center row of both inputs (lowers to global_prefetch_b8) */
    if (tid < 64) {
        size_t po = ((size_t)b*NC)*HW + (size_t)y*NW + x0 + tid;
        __builtin_prefetch(&inA[po], 0, 0);
        __builtin_prefetch(&inB[po], 0, 0);
    }

    /* weights: sW[t][n][c] = w[n][c][t], zero padded */
    for (int idx = tid; idx < 9*16*32; idx += 128) {
        int t   = idx >> 9;          /* /(16*32) */
        int rem = idx & 511;
        int n   = rem >> 5;
        int c   = rem & 31;
        float v = 0.0f;
        if (n < 10 && c < 20) v = wgt[(n*20 + c)*9 + t];
        sW[t][n][c] = (_Float16)v;
    }

    /* inputs: sIn[r][li][c], gx = x0-1+li, channel-contiguous writes */
    for (int site = tid; site < 3*66; site += 128) {
        int r  = site / 66;
        int li = site - r*66;
        int gx = x0 - 1 + li;
        int gy = y + r - 1;
        bool inb = (gx >= 0 && gx < NW && gy >= 0 && gy < NH);
        float mA = 1.0f, mB = 1.0f;
        size_t pbase = 0;
        if (inb) {
            pbase = ((size_t)b*NC)*HW + (size_t)gy*NW + gx;
            int mi = (b*NH + gy)*NW + gx;
            if (modA) mA = modA[mi];
            if (modB) mB = modB[mi];
        }
        _Float16* dst = &sIn[r][li][0];
#pragma unroll
        for (int c = 0; c < 10; ++c)
            dst[c] = (_Float16)(inb ? inA[pbase + (size_t)c*HW] * mA : 0.0f);
#pragma unroll
        for (int c = 0; c < 10; ++c)
            dst[10 + c] = (_Float16)(inb ? inB[pbase + (size_t)c*HW] * mB : 0.0f);
#pragma unroll
        for (int c = 20; c < 32; ++c)
            dst[c] = (_Float16)0.0f;
    }
    __syncthreads();

    int lane = tid & 31;
    int wv   = tid >> 5;        /* wave 0..3: 16-pixel tile each */
    int mlo  = lane & 15;
    int mh   = lane >> 4;
    int xx0  = wv*16 + mlo;     /* pixel row index of this lane in sIn (before dx) */

    v8f acc = {};
#pragma unroll
    for (int dy = 0; dy < 3; ++dy) {
#pragma unroll
        for (int dx = 0; dx < 3; ++dx) {
            int t = dy*3 + dx;
            /* A (16x32 f16): lane needs K = mh*8+0..7 and 16+mh*8+0..7 -> 2x b128 */
            const v8h* ap = (const v8h*)&sIn[dy][xx0 + dx][mh*8];
            v8h a0 = ap[0];
            v8h a1 = ap[2];                       /* +16 halves */
            /* B (32x16 f16): lane needs K = mh*8+0..7 and 16+mh*8+0..7 at column mlo -> 2x b128 */
            const v8h* bp = (const v8h*)&sW[t][mlo][mh*8];
            v8h b0 = bp[0];
            v8h b1 = bp[2];
            v16h A  = __builtin_shufflevector(a0, a1, 0,1,2,3,4,5,6,7,8,9,10,11,12,13,14,15);
            v16h Bf = __builtin_shufflevector(b0, b1, 0,1,2,3,4,5,6,7,8,9,10,11,12,13,14,15);
            acc = __builtin_amdgcn_wmma_f32_16x16x32_f16(false, A, false, Bf,
                                                         (short)0, acc, false, false);
        }
    }

    /* D layout: n = lane&15, m = vgpr + 8*(lane>>4) */
    if (mlo < 10) {
        float gg = gam[mlo], bb = bet[mlo];
#pragma unroll
        for (int r8 = 0; r8 < 8; ++r8) {
            int m = r8 + mh*8;
            int x = x0 + wv*16 + m;
            float v = acc[r8] * gg + bb;
            v = v > 0.0f ? v : 0.0f;
            size_t o = ((size_t)(b*NC + mlo))*HW + (size_t)y*NW + x;
            if (accumulate) outp[o] += v;
            else            outp[o]  = v;
        }
    }
}

/* ------------------------------------------------------------------ */
/* Kernel 3: ConvGRU (1x1 gates + candidate + blend), per pixel        */
/* ------------------------------------------------------------------ */
__global__ __launch_bounds__(256) void gru_kernel(
    const float* __restrict__ msg, const float* __restrict__ h,
    const float* __restrict__ wg, const float* __restrict__ bg,
    const float* __restrict__ wc, const float* __restrict__ g,
    const float* __restrict__ be, float* __restrict__ outslot)
{
    __shared__ float s_wg[40], s_bg[2], s_wc[200], s_g[10], s_be[10];
    int tid = threadIdx.x;
    if (tid < 40)  s_wg[tid] = wg[tid];
    if (tid < 2)   s_bg[tid] = bg[tid];
    if (tid < 200) s_wc[tid] = wc[tid];
    if (tid < 10)  { s_g[tid] = g[tid]; s_be[tid] = be[tid]; }
    __syncthreads();

    int pix = blockIdx.x * 256 + tid;
    int b  = pix / HW;
    int hw = pix - b*HW;
    size_t base = (size_t)b*NC*HW + hw;

    float mv[10], hv[10];
#pragma unroll
    for (int c = 0; c < 10; ++c) {
        mv[c] = msg[base + (size_t)c*HW];
        hv[c] = h  [base + (size_t)c*HW];
    }

    float gate[2];
#pragma unroll
    for (int o = 0; o < 2; ++o) {
        float s = s_bg[o];
#pragma unroll
        for (int c = 0; c < 10; ++c)
            s += s_wg[o*20 + c]*mv[c] + s_wg[o*20 + 10 + c]*hv[c];
        gate[o] = 1.0f / (1.0f + __expf(-s));
    }
    float r = gate[0], u = gate[1];

#pragma unroll
    for (int o = 0; o < 10; ++o) {
        float s = 0.0f;
#pragma unroll
        for (int c = 0; c < 10; ++c)
            s += s_wc[o*20 + c]*mv[c] + s_wc[o*20 + 10 + c]*(r*hv[c]);
        s = s_g[o]*s + s_be[o];
        s = s > 0.0f ? s : 0.01f*s;            /* leaky_relu 0.01 */
        outslot[base + (size_t)o*HW] = (1.0f - u)*hv[o] + u*s;
    }
}

/* ------------------------------------------------------------------ */
extern "C" void kernel_launch(void* const* d_in, const int* in_sizes, int n_in,
                              void* d_out, int out_size, void* d_ws, size_t ws_size,
                              hipStream_t stream)
{
    (void)in_sizes; (void)n_in; (void)out_size; (void)ws_size;

    const float* f_nodes  = (const float*)d_in[0];
    const float* h_nodes  = (const float*)d_in[1];
    const float* p_nodes  = (const float*)d_in[2];
    const float* w_dmap   = (const float*)d_in[4];
    const float* b_dmap   = (const float*)d_in[5];
    const float* w_decomp = (const float*)d_in[6];
    const float* g_decomp = (const float*)d_in[7];
    const float* be_decomp= (const float*)d_in[8];
    const float* w_cau    = (const float*)d_in[9];
    const float* b_cau    = (const float*)d_in[10];
    const float* w_cal    = (const float*)d_in[11];
    const float* b_cal    = (const float*)d_in[12];
    const float* w_cu     = (const float*)d_in[13];
    const float* g_cu     = (const float*)d_in[14];
    const float* be_cu    = (const float*)d_in[15];
    const float* w_cl     = (const float*)d_in[16];
    const float* g_cl     = (const float*)d_in[17];
    const float* be_cl    = (const float*)d_in[18];
    const float* wg_u     = (const float*)d_in[19];
    const float* bg_u     = (const float*)d_in[20];
    const float* wc_u     = (const float*)d_in[21];
    const float* g_u      = (const float*)d_in[22];
    const float* be_u     = (const float*)d_in[23];
    const float* wg_l     = (const float*)d_in[24];
    const float* bg_l     = (const float*)d_in[25];
    const float* wc_l     = (const float*)d_in[26];
    const float* g_l      = (const float*)d_in[27];
    const float* be_l     = (const float*)d_in[28];

    float* outp = (float*)d_out;
    float* ws   = (float*)d_ws;

    const float* f1 = f_nodes + (size_t)BCHW;
    const float* h0 = h_nodes;
    const float* h1 = h_nodes + (size_t)BCHW;
    const float* h2 = h_nodes + (size_t)2*BCHW;

    fuse_maps_kernel<<<BHW/256, 256, 0, stream>>>(
        f1, h0, h1, h2, p_nodes,
        w_dmap, b_dmap, w_cau, b_cau, w_cal, b_cal, outp, ws);

    dim3 cgrid(NB*NH*3), cblk(128);

    /* msg_u = d0 + sum_i relu(bn(conv([h1, p_i*cmu]))) */
    conv3x3_wmma_kernel<<<cgrid, cblk, 0, stream>>>(
        f1, ws + WS_ATT1, h1, nullptr, w_decomp, g_decomp, be_decomp,
        ws + WS_MSG_U, 0);
    for (int i = 1; i <= 4; ++i)
        conv3x3_wmma_kernel<<<cgrid, cblk, 0, stream>>>(
            h1, nullptr, p_nodes + (size_t)i*BCHW, outp + OUT_CMU,
            w_cu, g_cu, be_cu, ws + WS_MSG_U, 1);

    /* msg_l = d1 + sum_i relu(bn(conv([h2, p_i*cml]))) */
    conv3x3_wmma_kernel<<<cgrid, cblk, 0, stream>>>(
        f1, ws + WS_ATT2, h2, nullptr, w_decomp, g_decomp, be_decomp,
        ws + WS_MSG_L, 0);
    for (int i = 5; i <= 6; ++i)
        conv3x3_wmma_kernel<<<cgrid, cblk, 0, stream>>>(
            h2, nullptr, p_nodes + (size_t)i*BCHW, outp + OUT_CML,
            w_cl, g_cl, be_cl, ws + WS_MSG_L, 1);

    /* GRUs -> xh_new slots 1 and 2 */
    gru_kernel<<<BHW/256, 256, 0, stream>>>(
        ws + WS_MSG_U, h1, wg_u, bg_u, wc_u, g_u, be_u, outp + (size_t)BCHW);
    gru_kernel<<<BHW/256, 256, 0, stream>>>(
        ws + WS_MSG_L, h2, wg_l, bg_l, wc_l, g_l, be_l, outp + (size_t)2*BCHW);
}
